// NonLocalBlock2D_17317308137919
// MI455X (gfx1250) — compile-verified, matrix operand order checked by reading, not run on hardware
//
#include <hip/hip_runtime.h>
#include <hip/hip_bf16.h>

// ---------------------------------------------------------------------------
// NonLocalBlock2D fused pipeline for MI455X (gfx1250, wave32, WMMA bf16)
// B=8, C=256, O=128, H=W=64 -> N=4096, M=1024
// Fragments: 2 x ds_load_b128 per operand. Staging: 4-wide vector loads.
// ---------------------------------------------------------------------------

typedef __attribute__((ext_vector_type(16))) __bf16 v16bf;
typedef __attribute__((ext_vector_type(8)))  __bf16 v8bf;
typedef __attribute__((ext_vector_type(4)))  __bf16 v4bf;
typedef __attribute__((ext_vector_type(8)))  float  v8f;
typedef __attribute__((ext_vector_type(4)))  float  v4f;

constexpr int kB = 8;
constexpr int kC = 256;
constexpr int kO = 128;
constexpr int kH = 64;
constexpr int kW = 64;
constexpr int kN = kH * kW;        // 4096
constexpr int kM = (kH/2)*(kW/2);  // 1024

__device__ __forceinline__ v8f wmma_bf16(v16bf a, v16bf b, v8f c) {
  return __builtin_amdgcn_wmma_f32_16x16x32_bf16(false, a, false, b,
                                                 (short)0, c, false, false);
}

// aligned vector ld/st helpers
__device__ __forceinline__ v8bf ld8(const __bf16* p) {
  v8bf r; __builtin_memcpy(&r, __builtin_assume_aligned(p, 16), 16); return r;
}
__device__ __forceinline__ v4bf ld4(const __bf16* p) {
  v4bf r; __builtin_memcpy(&r, __builtin_assume_aligned(p, 8), 8); return r;
}
__device__ __forceinline__ void st4(__bf16* p, v4bf v) {
  __builtin_memcpy(__builtin_assume_aligned(p, 8), &v, 8);
}
__device__ __forceinline__ v4f ld4f(const float* p) {
  v4f r; __builtin_memcpy(&r, __builtin_assume_aligned(p, 16), 16); return r;
}
__device__ __forceinline__ v4bf cvt4(v4f v) {
  v4bf r;
  #pragma unroll
  for (int j = 0; j < 4; ++j) r[j] = (__bf16)v[j];   // RNE hw convert
  return r;
}

// CDNA5 16-bit A/B fragment from a K-contiguous row (base = &row[k0]):
// lanes 0-15 take K {0..7, 16..23}, lanes 16-31 take K {8..15, 24..31}.
__device__ __forceinline__ v16bf load_frag_row(const __bf16* row, int lane) {
  const int kb = (lane & 16) ? 8 : 0;
  v8bf lo = ld8(row + kb);
  v8bf hi = ld8(row + kb + 16);
  return __builtin_shufflevector(lo, hi, 0,1,2,3,4,5,6,7,8,9,10,11,12,13,14,15);
}

// Row strides (elements) keeping every 8-element K-run 16B aligned.
constexpr int kP32  = 40;   // K-width 32  (80 B rows)
constexpr int kP128 = 136;  // K-width 128 (272 B rows)

// ---------------------------------------------------------------------------
// Kernel 1: projection GEMM  out[b,o,n] = sum_c W[o,c]*x[b,c,n] + bias[o]
// Block: 256 thr (8 waves). Tile: O=128 x N=64, K-steps of 32 over C=256.
// ---------------------------------------------------------------------------
__global__ __launch_bounds__(256)
void proj_gemm_kernel(const float* __restrict__ x,
                      const float* __restrict__ w,
                      const float* __restrict__ bias,
                      __bf16* __restrict__ out) {
  const int b   = blockIdx.y;
  const int n0  = blockIdx.x * 64;
  const int tid = threadIdx.x;
  const int wave = tid >> 5, lane = tid & 31;

  __shared__ alignas(16) __bf16 lds_w[128][kP32];   // [o][k]   A: K-contig
  __shared__ alignas(16) __bf16 lds_xT[64][kP32];   // [n][k]   B: K-contig

  v8f acc[4] = {};
  const float* xb = x + (size_t)b * kC * kN;

  for (int c0 = 0; c0 < kC; c0 += 32) {
    // W tile 128x32: float4 loads, packed b64 LDS stores
    for (int i = tid; i < 1024; i += 256) {
      const int o = i >> 3, c4 = (i & 7) * 4;
      st4(&lds_w[o][c4], cvt4(ld4f(&w[(size_t)o * kC + c0 + c4])));
    }
    // X tile 32x64 (transposed into [n][k]): float4 loads, 4x b16 stores
    for (int i = tid; i < 512; i += 256) {
      const int cc = i >> 4, n4 = (i & 15) * 4;
      const v4f xv = ld4f(&xb[(size_t)(c0 + cc) * kN + n0 + n4]);
      #pragma unroll
      for (int j = 0; j < 4; ++j) lds_xT[n4 + j][cc] = (__bf16)xv[j];
      if (c0 + 32 < kC)   // hide HBM latency of the next K-tile
        __builtin_prefetch(&xb[(size_t)(c0 + 32 + cc) * kN + n0 + n4], 0, 3);
    }
    __syncthreads();

    const v16bf a = load_frag_row(&lds_w[wave * 16 + (lane & 15)][0], lane);
    #pragma unroll
    for (int j = 0; j < 4; ++j) {
      const v16bf bf = load_frag_row(&lds_xT[j * 16 + (lane & 15)][0], lane);
      acc[j] = wmma_bf16(a, bf, acc[j]);
    }
    __syncthreads();
  }

  #pragma unroll
  for (int j = 0; j < 4; ++j) {
    #pragma unroll
    for (int r = 0; r < 8; ++r) {
      const int o = wave * 16 + r + ((lane & 16) ? 8 : 0);
      const int n = n0 + j * 16 + (lane & 15);
      out[((size_t)b * kO + o) * kN + n] = (__bf16)(acc[j][r] + bias[o]);
    }
  }
}

// ---------------------------------------------------------------------------
// Kernel 2: 2x2 maxpool, bf16 -> bf16, 2 outputs / thread (b64 in, b32 out)
// ---------------------------------------------------------------------------
__global__ void maxpool_kernel(const __bf16* __restrict__ in,
                               __bf16* __restrict__ out) {
  const int idx = blockIdx.x * blockDim.x + threadIdx.x;
  if (idx >= kB * kO * 32 * 16) return;
  const int wo2 = idx & 15, ho = (idx >> 4) & 31, ch = idx >> 9;
  const __bf16* p = in + ((size_t)ch * 64 + ho * 2) * 64 + wo2 * 4;
  const v4bf r0 = ld4(p), r1 = ld4(p + 64);
  const float m0 = fmaxf(fmaxf((float)r0[0], (float)r0[1]),
                         fmaxf((float)r1[0], (float)r1[1]));
  const float m1 = fmaxf(fmaxf((float)r0[2], (float)r0[3]),
                         fmaxf((float)r1[2], (float)r1[3]));
  __bf16 o2[2] = { (__bf16)m0, (__bf16)m1 };
  __builtin_memcpy(out + (size_t)ch * 1024 + ho * 32 + wo2 * 2, o2, 4);
}

// ---------------------------------------------------------------------------
// Kernel 3: fused flash attention (never materializes [N,M]).
// Block: 256 thr, 8 waves; wave owns 16 queries (block covers 128).
// Softmax stats register-resident via __shfl_xor; LDS only for operand
// tiles and the P transpose.  Output stored TRANSPOSED: y[b, n, o].
// ---------------------------------------------------------------------------
__global__ __launch_bounds__(256)
void attn_kernel(const __bf16* __restrict__ theta,
                 const __bf16* __restrict__ phi,
                 const __bf16* __restrict__ g,
                 __bf16* __restrict__ yT) {
  const int b    = blockIdx.y;
  const int n0   = blockIdx.x * 128;
  const int tid  = threadIdx.x;
  const int wave = tid >> 5, lane = tid & 31;
  const int qbase = n0 + wave * 16;
  const int half = (lane & 16) ? 8 : 0;

  __shared__ alignas(16) __bf16 lds_thT[128][kP128];  // [q][o]  34.8 KB
  __shared__ alignas(16) __bf16 lds_phiT[32][kP128];  // [m][o]   8.7 KB
  __shared__ alignas(16) __bf16 lds_g[128][kP32];     // [o][m]  10.2 KB
  __shared__ alignas(16) __bf16 lds_p[8][16][kP32];   // per-wave P  10.2 KB

  // ---- stage theta^T tile [128 q][128 o] once per block (b64 loads) ----
  for (int i = tid; i < 4096; i += 256) {
    const int o = i >> 5, n4 = (i & 31) * 4;
    const v4bf t = ld4(&theta[((size_t)b * kO + o) * kN + n0 + n4]);
    #pragma unroll
    for (int j = 0; j < 4; ++j) lds_thT[n4 + j][o] = t[j];
  }
  __syncthreads();

  // theta A-fragments: rows = 16 queries, K = O (4 chunks of 32)
  v16bf ath[4];
  #pragma unroll
  for (int kc = 0; kc < 4; ++kc)
    ath[kc] = load_frag_row(&lds_thT[wave * 16 + (lane & 15)][kc * 32], lane);

  v8f yacc[8] = {};
  float rmax[8], rsum[8];
  #pragma unroll
  for (int r = 0; r < 8; ++r) { rmax[r] = -1e30f; rsum[r] = 0.f; }

  for (int m0 = 0; m0 < kM; m0 += 32) {
    __syncthreads();   // previous iteration readers done
    for (int i = tid; i < 1024; i += 256) {
      const int o = i >> 3, m4 = (i & 7) * 4;
      const size_t src = ((size_t)b * kO + o) * kM + m0 + m4;
      const v4bf pv = ld4(&phi[src]);
      #pragma unroll
      for (int j = 0; j < 4; ++j) lds_phiT[m4 + j][o] = pv[j];
      st4(&lds_g[o][m4], ld4(&g[src]));
    }
    __syncthreads();

    // ---- S tile: 16 queries x 32 keys, K over O=128 ----
    v8f s0 = {}, s1 = {};
    #pragma unroll
    for (int kc = 0; kc < 4; ++kc) {
      const v16bf bp0 = load_frag_row(&lds_phiT[lane & 15][kc * 32], lane);
      const v16bf bp1 = load_frag_row(&lds_phiT[16 + (lane & 15)][kc * 32], lane);
      s0 = wmma_bf16(ath[kc], bp0, s0);
      s1 = wmma_bf16(ath[kc], bp1, s1);
    }

    // ---- online softmax, all in registers ----
    // Row q = r + half lives across this lane's 16-lane group: reduce with
    // xor-shuffles (masks < 16 stay inside the group on wave32).
    #pragma unroll
    for (int r = 0; r < 8; ++r) {
      float mx = fmaxf(s0[r], s1[r]);
      #pragma unroll
      for (int off = 8; off; off >>= 1)
        mx = fmaxf(mx, __shfl_xor(mx, off, 32));
      const float mnew = fmaxf(rmax[r], mx);
      const float al = __expf(rmax[r] - mnew);
      const float p0 = __expf(s0[r] - mnew);
      const float p1 = __expf(s1[r] - mnew);
      float ps = p0 + p1;
      #pragma unroll
      for (int off = 8; off; off >>= 1)
        ps += __shfl_xor(ps, off, 32);
      rmax[r] = mnew;
      rsum[r] = rsum[r] * al + ps;
      #pragma unroll
      for (int j = 0; j < 8; ++j) yacc[j][r] *= al;   // rescale running Y
      // P into per-wave LDS in A layout (row q, K-contiguous keys)
      const int q = r + half;
      lds_p[wave][q][lane & 15]        = (__bf16)p0;
      lds_p[wave][q][16 + (lane & 15)] = (__bf16)p1;
    }
    asm volatile("s_wait_dscnt 0" ::: "memory");  // same-wave LDS RAW

    const v16bf pa = load_frag_row(&lds_p[wave][lane & 15][0], lane);

    // ---- Y += P @ g^T : K = 32 keys, N = O (8 tiles of 16) ----
    #pragma unroll
    for (int j = 0; j < 8; ++j) {
      const v16bf bg = load_frag_row(&lds_g[j * 16 + (lane & 15)][0], lane);
      yacc[j] = wmma_bf16(pa, bg, yacc[j]);
    }
  }

  // epilogue: scale by 1/sum, store transposed y[b, n, o]
  #pragma unroll
  for (int r = 0; r < 8; ++r) {
    const int q = r + half;
    const float rinv = 1.f / rsum[r];
    #pragma unroll
    for (int j = 0; j < 8; ++j) {
      const int o = j * 16 + (lane & 15);
      yT[((size_t)b * kN + qbase + q) * kO + o] = (__bf16)(yacc[j][r] * rinv);
    }
  }
}

// ---------------------------------------------------------------------------
// Kernel 4: output GEMM + bias + BatchNorm(inference) + residual (fp32 out)
//   out[b,c,n] = (sum_o w_out[c,o]*yT[b,n,o] + b_out[c] - mean[c])*inv[c]
//                + beta[c] + x[b,c,n]
// yT is [B,N,O]: the B operand is K(=O)-contiguous -> no LDS transpose.
// ---------------------------------------------------------------------------
__global__ __launch_bounds__(256)
void out_gemm_kernel(const __bf16* __restrict__ yT,
                     const float* __restrict__ w_out,
                     const float* __restrict__ b_out,
                     const float* __restrict__ gamma,
                     const float* __restrict__ beta,
                     const float* __restrict__ mean,
                     const float* __restrict__ var,
                     const float* __restrict__ x,
                     float* __restrict__ out) {
  const int b     = blockIdx.y;
  const int n0    = blockIdx.x * 64;
  const int c_blk = blockIdx.z * 128;
  const int tid   = threadIdx.x;
  const int wave  = tid >> 5, lane = tid & 31;

  __shared__ alignas(16) __bf16 lds_w[128][kP32];  // [c][k]
  __shared__ alignas(16) __bf16 lds_y[64][kP32];   // [n][k]

  v8f acc[4] = {};
  const __bf16* ybb = yT + (size_t)b * kN * kO;

  for (int k0 = 0; k0 < kO; k0 += 32) {
    for (int i = tid; i < 1024; i += 256) {
      const int c = i >> 3, k4 = (i & 7) * 4;
      st4(&lds_w[c][k4],
          cvt4(ld4f(&w_out[(size_t)(c_blk + c) * kO + k0 + k4])));
    }
    for (int i = tid; i < 512; i += 256) {
      const int n = i >> 3, k4 = (i & 7) * 4;
      st4(&lds_y[n][k4], ld4(&ybb[(size_t)(n0 + n) * kO + k0 + k4]));
      if (k0 + 32 < kO)
        __builtin_prefetch(&ybb[(size_t)(n0 + n) * kO + k0 + 32 + k4], 0, 3);
    }
    __syncthreads();

    const v16bf a = load_frag_row(&lds_w[wave * 16 + (lane & 15)][0], lane);
    #pragma unroll
    for (int j = 0; j < 4; ++j) {
      const v16bf bf = load_frag_row(&lds_y[j * 16 + (lane & 15)][0], lane);
      acc[j] = wmma_bf16(a, bf, acc[j]);
    }
    __syncthreads();
  }

  #pragma unroll
  for (int r = 0; r < 8; ++r) {
    const int c = c_blk + wave * 16 + r + ((lane & 16) ? 8 : 0);
    const float inv = gamma[c] * rsqrtf(var[c] + 1e-5f);
    const float mu = mean[c], bt = beta[c], bo = b_out[c];
    #pragma unroll
    for (int j = 0; j < 4; ++j) {
      const int n = n0 + j * 16 + (lane & 15);
      const size_t off = ((size_t)b * kC + c) * kN + n;
      float v = acc[j][r] + bo;
      v = (v - mu) * inv + bt + x[off];
      out[off] = v;
    }
  }
}

// ---------------------------------------------------------------------------
extern "C" void kernel_launch(void* const* d_in, const int* in_sizes, int n_in,
                              void* d_out, int out_size, void* d_ws, size_t ws_size,
                              hipStream_t stream) {
  const float* x        = (const float*)d_in[0];
  const float* w_theta  = (const float*)d_in[1];
  const float* b_theta  = (const float*)d_in[2];
  const float* w_phi    = (const float*)d_in[3];
  const float* b_phi    = (const float*)d_in[4];
  const float* w_g      = (const float*)d_in[5];
  const float* b_g      = (const float*)d_in[6];
  const float* w_out    = (const float*)d_in[7];
  const float* b_out    = (const float*)d_in[8];
  const float* bn_gamma = (const float*)d_in[9];
  const float* bn_beta  = (const float*)d_in[10];
  const float* bn_mean  = (const float*)d_in[11];
  const float* bn_var   = (const float*)d_in[12];
  float* out = (float*)d_out;

  // workspace layout (bf16 buffers), ~36 MB total
  char* ws = (char*)d_ws;
  const size_t MB = 1024u * 1024u;
  __bf16* theta = (__bf16*)(ws);              // [B,O,N]  8 MB
  __bf16* phiF  = (__bf16*)(ws + 8 * MB);     // [B,O,N]  8 MB
  __bf16* gF    = (__bf16*)(ws + 16 * MB);    // [B,O,N]  8 MB
  __bf16* phiP  = (__bf16*)(ws + 24 * MB);    // [B,O,M]  2 MB
  __bf16* gP    = (__bf16*)(ws + 26 * MB);    // [B,O,M]  2 MB
  __bf16* ybufT = (__bf16*)(ws + 28 * MB);    // [B,N,O]  8 MB (transposed)

  const dim3 gproj(kN / 64, kB);
  proj_gemm_kernel<<<gproj, 256, 0, stream>>>(x, w_theta, b_theta, theta);
  proj_gemm_kernel<<<gproj, 256, 0, stream>>>(x, w_phi,   b_phi,   phiF);
  proj_gemm_kernel<<<gproj, 256, 0, stream>>>(x, w_g,     b_g,     gF);

  const int pool_n = kB * kO * 32 * 16;
  maxpool_kernel<<<(pool_n + 255) / 256, 256, 0, stream>>>(phiF, phiP);
  maxpool_kernel<<<(pool_n + 255) / 256, 256, 0, stream>>>(gF, gP);

  attn_kernel<<<dim3(kN / 128, kB), 256, 0, stream>>>(theta, phiP, gP, ybufT);

  out_gemm_kernel<<<dim3(kN / 64, kB, kC / 128), 256, 0, stream>>>(
      ybufT, w_out, b_out, bn_gamma, bn_beta, bn_mean, bn_var, x, out);
}
